// Ground_att_81913616269994
// MI455X (gfx1250) — matrix operand order repaired and checked by reading, not compiled, not run on hardware
//
#include <hip/hip_runtime.h>
#include <hip/hip_bf16.h>

typedef __bf16 bf16_t;
typedef __attribute__((ext_vector_type(16))) __bf16 v16bf;
typedef __attribute__((ext_vector_type(8)))  __bf16 v8bf;
typedef __attribute__((ext_vector_type(8)))  float  v8f;

#define WMMA_BF16(a, b, c) \
  __builtin_amdgcn_wmma_f32_16x16x32_bf16(false, (a), false, (b), (short)0, (c), false, false)

__device__ __forceinline__ bf16_t to_bf16(float f) {
  unsigned u = __float_as_uint(f);
  unsigned r = (u + 0x7FFFu + ((u >> 16) & 1u)) >> 16;
  unsigned short s = (unsigned short)r;
  return __builtin_bit_cast(bf16_t, s);
}
__device__ __forceinline__ float bf16_to_f(bf16_t h) {
  unsigned short s = __builtin_bit_cast(unsigned short, h);
  return __uint_as_float(((unsigned)s) << 16);
}

// A fragment (16x32, MxK): lane&15 = row; lanes 0-15 hold K[0..15],
// lanes 16-31 hold K[16..31] (contiguous 16).
__device__ __forceinline__ v16bf load_a_frag(const bf16_t* base, int lane, int stride) {
  const bf16_t* p = base + (size_t)(lane & 15) * (size_t)stride + ((lane >> 4) << 4);
  v8bf lo = *(const v8bf*)p;
  v8bf hi = *(const v8bf*)(p + 8);
  return __builtin_shufflevector(lo, hi, 0, 1, 2, 3, 4, 5, 6, 7, 8, 9, 10, 11, 12, 13, 14, 15);
}
// B fragment (32x16, KxN) loaded from Bt[N][K] row-major: lane&15 = column (row
// of Bt); lanes 0-15 hold K[0..15], lanes 16-31 hold K[16..31] (contiguous 16).
__device__ __forceinline__ v16bf load_b_frag(const bf16_t* base, int lane, size_t stride) {
  const bf16_t* p = base + (size_t)(lane & 15) * stride + ((lane >> 4) << 4);
  return *(const v16bf*)p;
}

#define GEMM_STORE_BF16 0
#define GEMM_STORE_VT   1
#define GEMM_STORE_F32  2
#define GEMM_RELU_BF16  3

// Block = 256 threads = 8 waves arranged 2(M) x 4(N); block tile 64(M) x 256(N);
// wave tile 32 x 64: 2 A frags feed 8 WMMAs per K-step (~21 FLOP/B of L1 traffic).
__global__ __launch_bounds__(256)
void gemm_bf16_kernel(const bf16_t* __restrict__ A, int lda,
                      const bf16_t* __restrict__ Bt, int K,
                      void* __restrict__ Cout, int N,
                      int mode, int Srows) {
  const int t = threadIdx.x, wave = t >> 5, lane = t & 31;
  const int m0 = blockIdx.y * 64 + (wave >> 2) * 32;
  const int n0 = blockIdx.x * 256 + (wave & 3) * 64;

  v8f acc[2][4];
#pragma unroll
  for (int i = 0; i < 2; ++i)
#pragma unroll
    for (int j = 0; j < 4; ++j)
#pragma unroll
      for (int r = 0; r < 8; ++r) acc[i][j][r] = 0.f;

  const bf16_t* Abase = A + (size_t)m0 * (size_t)lda;
  const bf16_t* B0 = Bt + (size_t)n0 * (size_t)K;
  for (int k0 = 0; k0 < K; k0 += 32) {
    v16bf a0 = load_a_frag(Abase + k0, lane, lda);
    v16bf a1 = load_a_frag(Abase + (size_t)16 * (size_t)lda + k0, lane, lda);
#pragma unroll
    for (int j = 0; j < 4; ++j) {
      v16bf b = load_b_frag(B0 + (size_t)(j * 16) * (size_t)K + k0, lane, (size_t)K);
      acc[0][j] = WMMA_BF16(a0, b, acc[0][j]);
      acc[1][j] = WMMA_BF16(a1, b, acc[1][j]);
    }
  }

  const int hi = lane >> 4, col = lane & 15;
#pragma unroll
  for (int i = 0; i < 2; ++i) {
#pragma unroll
    for (int j = 0; j < 4; ++j) {
#pragma unroll
      for (int r = 0; r < 8; ++r) {
        int row = m0 + i * 16 + r + 8 * hi;
        int cc = n0 + j * 16 + col;
        float v = acc[i][j][r];
        if (mode == GEMM_STORE_BF16) {
          ((bf16_t*)Cout)[(size_t)row * N + cc] = to_bf16(v);
        } else if (mode == GEMM_STORE_VT) {
          int nb = row / Srows, s = row % Srows;
          ((bf16_t*)Cout)[((size_t)nb * N + cc) * (size_t)Srows + s] = to_bf16(v);
        } else if (mode == GEMM_STORE_F32) {
          ((float*)Cout)[(size_t)row * N + cc] = v;
        } else {
          v = v > 0.f ? v : 0.f;
          ((bf16_t*)Cout)[(size_t)row * N + cc] = to_bf16(v);
        }
      }
    }
  }
}

// Fused attention: one block per (n, 16-row L tile). 8 waves; each wave owns a
// 16x64 slice of the D=512 output accumulator. Online softmax over S in blocks
// of 128 columns. Softmax of +distance, exactly as the reference.
__global__ __launch_bounds__(256)
void attn_kernel(const bf16_t* __restrict__ qb, const bf16_t* __restrict__ kb,
                 const bf16_t* __restrict__ vT, const float* __restrict__ q2,
                 const float* __restrict__ k2, bf16_t* __restrict__ msg,
                 int L, int S, int D) {
  __shared__ bf16_t qs[16 * 512];
  __shared__ float sc[16][128];
  __shared__ bf16_t Ps[16 * 128];
  __shared__ float pred[16][16];
  __shared__ float mrow[16], lrow[16], arow[16], q2s[16];

  const int n = blockIdx.y;
  const int l0 = blockIdx.x * 16;
  const int t = threadIdx.x, wave = t >> 5, lane = t & 31;
  const int hi = lane >> 4, col = lane & 15;
  const int dw = wave * 64;

  // stage q tile (16 x 512 bf16 = 16KB) into LDS
  {
    const uint4* src4 = (const uint4*)(qb + ((size_t)n * L + l0) * (size_t)D);
    uint4* dst4 = (uint4*)qs;
#pragma unroll
    for (int i = 0; i < 4; ++i) dst4[t + i * 256] = src4[t + i * 256];
  }
  if (t < 16) {
    q2s[t] = q2[(size_t)n * L + l0 + t];
    mrow[t] = -1e30f;
    lrow[t] = 0.f;
  }
  __syncthreads();

  v8f acc[4];
#pragma unroll
  for (int j = 0; j < 4; ++j)
#pragma unroll
    for (int r = 0; r < 8; ++r) acc[j][r] = 0.f;

  for (int s0 = 0; s0 < S; s0 += 128) {
    // ---- phase 1: scores (each wave does a 16x16 column slab over D) ----
    v8f cs;
#pragma unroll
    for (int r = 0; r < 8; ++r) cs[r] = 0.f;
    const bf16_t* kbase = kb + ((size_t)n * S + s0 + wave * 16) * (size_t)D;
    for (int k0 = 0; k0 < D; k0 += 32) {
      v16bf a = load_a_frag(qs + k0, lane, D);
      v16bf b = load_b_frag(kbase + k0, lane, (size_t)D);
      cs = WMMA_BF16(a, b, cs);
    }
    const float k2v = k2[(size_t)n * S + s0 + wave * 16 + col];
#pragma unroll
    for (int r = 0; r < 8; ++r) {
      int row = r + 8 * hi;
      float dv = q2s[row] + k2v - 2.0f * cs[r];
      sc[row][wave * 16 + col] = sqrtf(fmaxf(dv, 0.f));
    }
    __syncthreads();

    // ---- phase 2a: block row max ----
    {
      int row = t & 15, c0 = (t >> 4) * 8;
      float m = -1e30f;
      for (int c = c0; c < c0 + 8; ++c) m = fmaxf(m, sc[row][c]);
      pred[row][t >> 4] = m;
    }
    __syncthreads();
    if (t < 16) {
      float bm = -1e30f;
      for (int i = 0; i < 16; ++i) bm = fmaxf(bm, pred[t][i]);
      float nm = fmaxf(mrow[t], bm);
      arow[t] = __expf(mrow[t] - nm);
      mrow[t] = nm;
    }
    __syncthreads();
    // ---- phase 2b: P = exp(sc - m), partial sums ----
    {
      int row = t & 15, c0 = (t >> 4) * 8;
      float nm = mrow[row], s = 0.f;
      for (int c = c0; c < c0 + 8; ++c) {
        float p = __expf(sc[row][c] - nm);
        Ps[row * 128 + c] = to_bf16(p);
        s += p;
      }
      pred[row][t >> 4] = s;
    }
    __syncthreads();
    if (t < 16) {
      float s = 0.f;
      for (int i = 0; i < 16; ++i) s += pred[t][i];
      lrow[t] = lrow[t] * arow[t] + s;
    }
    __syncthreads();

    // ---- rescale accumulators, acc += P @ vT-slice ----
    float al[8];
#pragma unroll
    for (int r = 0; r < 8; ++r) al[r] = arow[r + 8 * hi];
#pragma unroll
    for (int j = 0; j < 4; ++j)
#pragma unroll
      for (int r = 0; r < 8; ++r) acc[j][r] *= al[r];

    const bf16_t* vbase = vT + ((size_t)n * D + dw) * (size_t)S + s0;
#pragma unroll
    for (int kk = 0; kk < 4; ++kk) {
      v16bf a = load_a_frag(Ps + kk * 32, lane, 128);
#pragma unroll
      for (int j = 0; j < 4; ++j) {
        v16bf b = load_b_frag(vbase + (size_t)(j * 16) * (size_t)S + kk * 32, lane, (size_t)S);
        acc[j] = WMMA_BF16(a, b, acc[j]);
      }
    }
    __syncthreads();
  }

  // ---- finalize: divide by softmax denominator, write bf16 message ----
  float li[8];
#pragma unroll
  for (int r = 0; r < 8; ++r) li[r] = lrow[r + 8 * hi];
  bf16_t* mbase = msg + ((size_t)n * L + l0) * (size_t)D;
#pragma unroll
  for (int j = 0; j < 4; ++j)
#pragma unroll
    for (int r = 0; r < 8; ++r) {
      int row = r + 8 * hi;
      mbase[(size_t)row * D + dw + j * 16 + col] = to_bf16(acc[j][r] / li[r]);
    }
}

__global__ __launch_bounds__(256)
void rowsumsq_kernel(const bf16_t* __restrict__ X, float* __restrict__ out) {
  const int row = blockIdx.x, t = threadIdx.x;
  __shared__ float red[256];
  float a = bf16_to_f(X[(size_t)row * 512 + t]);
  float c = bf16_to_f(X[(size_t)row * 512 + t + 256]);
  red[t] = a * a + c * c;
  __syncthreads();
  for (int o = 128; o > 0; o >>= 1) {
    if (t < o) red[t] += red[t + o];
    __syncthreads();
  }
  if (t == 0) out[row] = red[0];
}

__global__ __launch_bounds__(256)
void ln_to_h_kernel(const float* __restrict__ Y, const float* __restrict__ g,
                    const float* __restrict__ b, bf16_t* __restrict__ h) {
  const int row = blockIdx.x, t = threadIdx.x;
  __shared__ float red[256];
  const float* y = Y + (size_t)row * 512;
  float v0 = y[t], v1 = y[t + 256];
  red[t] = v0 + v1;
  __syncthreads();
  for (int o = 128; o > 0; o >>= 1) {
    if (t < o) red[t] += red[t + o];
    __syncthreads();
  }
  float mu = red[0] / 512.f;
  __syncthreads();
  float d0 = v0 - mu, d1 = v1 - mu;
  red[t] = d0 * d0 + d1 * d1;
  __syncthreads();
  for (int o = 128; o > 0; o >>= 1) {
    if (t < o) red[t] += red[t + o];
    __syncthreads();
  }
  float rs = rsqrtf(red[0] / 512.f + 1e-5f);
  bf16_t* hr = h + (size_t)row * 1024 + 512;
  hr[t] = to_bf16(d0 * rs * g[t] + b[t]);
  hr[t + 256] = to_bf16(d1 * rs * g[t + 256] + b[t + 256]);
}

__global__ __launch_bounds__(256)
void ln_final_kernel(const float* __restrict__ Y, const float* __restrict__ x,
                     const float* __restrict__ g, const float* __restrict__ b,
                     float* __restrict__ out) {
  const int row = blockIdx.x, t = threadIdx.x;
  __shared__ float red[256];
  const float* y = Y + (size_t)row * 512;
  float v0 = y[t], v1 = y[t + 256];
  red[t] = v0 + v1;
  __syncthreads();
  for (int o = 128; o > 0; o >>= 1) {
    if (t < o) red[t] += red[t + o];
    __syncthreads();
  }
  float mu = red[0] / 512.f;
  __syncthreads();
  float d0 = v0 - mu, d1 = v1 - mu;
  red[t] = d0 * d0 + d1 * d1;
  __syncthreads();
  for (int o = 128; o > 0; o >>= 1) {
    if (t < o) red[t] += red[t + o];
    __syncthreads();
  }
  float rs = rsqrtf(red[0] / 512.f + 1e-5f);
  const float* xr = x + (size_t)row * 512;
  float* orow = out + (size_t)row * 512;
  orow[t] = xr[t] + d0 * rs * g[t] + b[t];
  orow[t + 256] = xr[t + 256] + d1 * rs * g[t + 256] + b[t + 256];
}

__global__ void cvt_x_to_h_kernel(const float* __restrict__ x, bf16_t* __restrict__ h,
                                  size_t total) {
  size_t i = (size_t)blockIdx.x * 256 + threadIdx.x;
  if (i >= total) return;
  size_t row = i >> 9;
  int c = (int)(i & 511);
  h[(row << 10) + c] = to_bf16(x[i]);
}

__global__ void cvt_f2bf_kernel(const float* __restrict__ s, bf16_t* __restrict__ d,
                                size_t total) {
  size_t i = (size_t)blockIdx.x * 256 + threadIdx.x;
  if (i < total) d[i] = to_bf16(s[i]);
}

// W[K][N] row-major fp32 -> Wt[N][K] row-major bf16
__global__ void transcvt_kernel(const float* __restrict__ W, bf16_t* __restrict__ Wt,
                                int K, int Nn) {
  size_t i = (size_t)blockIdx.x * 256 + threadIdx.x;
  size_t total = (size_t)K * (size_t)Nn;
  if (i >= total) return;
  int nn = (int)(i / K), kk = (int)(i % K);
  Wt[i] = to_bf16(W[(size_t)kk * Nn + nn]);
}

extern "C" void kernel_launch(void* const* d_in, const int* in_sizes, int n_in,
                              void* d_out, int out_size, void* d_ws, size_t ws_size,
                              hipStream_t stream) {
  (void)in_sizes; (void)n_in; (void)out_size; (void)ws_size;
  const int Nb = 8, L = 1024, S = 4096, D = 512;
  const int ML = Nb * L;   // 8192
  const int MS = Nb * S;   // 32768

  const float* x   = (const float*)d_in[0];
  const float* src = (const float*)d_in[1];
  const float* Wq  = (const float*)d_in[2];
  const float* Wk  = (const float*)d_in[3];
  const float* Wv  = (const float*)d_in[4];
  const float* Wm  = (const float*)d_in[5];
  const float* W1  = (const float*)d_in[6];
  const float* W2  = (const float*)d_in[7];
  const float* g1  = (const float*)d_in[8];
  const float* b1  = (const float*)d_in[9];
  const float* g2  = (const float*)d_in[10];
  const float* b2  = (const float*)d_in[11];
  float* out = (float*)d_out;

  char* ws = (char*)d_ws;
  size_t off = 0;
  auto alloc = [&](size_t bytes) -> void* {
    void* p = ws + off;
    off += (bytes + 255) & ~(size_t)255;
    return p;
  };

  bf16_t* h    = (bf16_t*)alloc((size_t)ML * 1024 * 2);  // [x | ln(msg@Wm)] bf16
  bf16_t* sb   = (bf16_t*)alloc((size_t)MS * D * 2);
  bf16_t* WqT  = (bf16_t*)alloc((size_t)D * D * 2);
  bf16_t* WkT  = (bf16_t*)alloc((size_t)D * D * 2);
  bf16_t* WvT  = (bf16_t*)alloc((size_t)D * D * 2);
  bf16_t* WmT  = (bf16_t*)alloc((size_t)D * D * 2);
  bf16_t* W1T  = (bf16_t*)alloc((size_t)1024 * 1024 * 2);
  bf16_t* W2T  = (bf16_t*)alloc((size_t)512 * 1024 * 2);
  bf16_t* qbuf = (bf16_t*)alloc((size_t)ML * D * 2);
  bf16_t* kbuf = (bf16_t*)alloc((size_t)MS * D * 2);
  bf16_t* vT   = (bf16_t*)alloc((size_t)Nb * D * S * 2);
  float*  q2   = (float*)alloc((size_t)ML * 4);
  float*  k2   = (float*)alloc((size_t)MS * 4);
  bf16_t* msg  = (bf16_t*)alloc((size_t)ML * D * 2);
  float*  y1   = (float*)alloc((size_t)ML * D * 4);
  bf16_t* hh   = (bf16_t*)alloc((size_t)ML * 1024 * 2);
  float*  y2   = (float*)alloc((size_t)ML * D * 4);

  // prep: conversions + weight transposes
  cvt_x_to_h_kernel<<<(ML * D) / 256, 256, 0, stream>>>(x, h, (size_t)ML * D);
  cvt_f2bf_kernel<<<(MS * D) / 256, 256, 0, stream>>>(src, sb, (size_t)MS * D);
  transcvt_kernel<<<(D * D) / 256, 256, 0, stream>>>(Wq, WqT, D, D);
  transcvt_kernel<<<(D * D) / 256, 256, 0, stream>>>(Wk, WkT, D, D);
  transcvt_kernel<<<(D * D) / 256, 256, 0, stream>>>(Wv, WvT, D, D);
  transcvt_kernel<<<(D * D) / 256, 256, 0, stream>>>(Wm, WmT, D, D);
  transcvt_kernel<<<(1024 * 1024) / 256, 256, 0, stream>>>(W1, W1T, 1024, 1024);
  transcvt_kernel<<<(1024 * 512) / 256, 256, 0, stream>>>(W2, W2T, 1024, 512);

  // projections: q = x@Wq (x = left half of h, lda=1024), k/v = source@Wk/Wv
  gemm_bf16_kernel<<<dim3(D / 256, ML / 64), 256, 0, stream>>>(
      h, 1024, WqT, D, qbuf, D, GEMM_STORE_BF16, 0);
  gemm_bf16_kernel<<<dim3(D / 256, MS / 64), 256, 0, stream>>>(
      sb, D, WkT, D, kbuf, D, GEMM_STORE_BF16, 0);
  gemm_bf16_kernel<<<dim3(D / 256, MS / 64), 256, 0, stream>>>(
      sb, D, WvT, D, vT, D, GEMM_STORE_VT, S);

  rowsumsq_kernel<<<ML, 256, 0, stream>>>(qbuf, q2);
  rowsumsq_kernel<<<MS, 256, 0, stream>>>(kbuf, k2);

  // fused distance-softmax attention -> msg (bf16)
  attn_kernel<<<dim3(L / 16, Nb), 256, 0, stream>>>(qbuf, kbuf, vT, q2, k2, msg, L, S, D);

  // msg @ Wm -> y1 (fp32); layernorm -> right half of h (bf16)
  gemm_bf16_kernel<<<dim3(D / 256, ML / 64), 256, 0, stream>>>(
      msg, D, WmT, D, y1, D, GEMM_STORE_F32, 0);
  ln_to_h_kernel<<<ML, 256, 0, stream>>>(y1, g1, b1, h);

  // MLP: relu(h@W1) @ W2 -> y2; out = x + LN(y2)
  gemm_bf16_kernel<<<dim3(1024 / 256, ML / 64), 256, 0, stream>>>(
      h, 1024, W1T, 1024, hh, 1024, GEMM_RELU_BF16, 0);
  gemm_bf16_kernel<<<dim3(D / 256, ML / 64), 256, 0, stream>>>(
      hh, 1024, W2T, 1024, y2, D, GEMM_STORE_F32, 0);
  ln_final_kernel<<<ML, 256, 0, stream>>>(y2, x, g2, b2, out);
}
